// ConditionalAutoregressiveSpline_70806830842471
// MI455X (gfx1250) — matrix operand order repaired and checked by reading, not compiled, not run on hardware
//
#include <hip/hip_runtime.h>

// ---------------- problem constants ----------------
#define BATCH   16384
#define N_IN    128
#define CTX     128
#define H       1024
#define NB      8
#define OUT_MULT 23                 // 3*NB-1
#define NOUT    (N_IN * OUT_MULT)   // 2944
#define KCAT    (N_IN + CTX)        // 256

typedef __attribute__((ext_vector_type(16))) __bf16 v16bf;
typedef __attribute__((ext_vector_type(8)))  float  v8f;

union Frag16 { v16bf v; uint4 q[2]; };

__device__ __forceinline__ unsigned short f2bf(float f) {
    union { float f; unsigned int u; } c; c.f = f;
    unsigned int u = c.u;
    unsigned int r = (u + 0x7FFFu + ((u >> 16) & 1u)) >> 16;
    return (unsigned short)r;
}

// CDNA5 async global->LDS copy (16B per lane), tracked by ASYNCcnt.
__device__ __forceinline__ void async_b128(unsigned lds_off, const void* gptr) {
    asm volatile("global_load_async_to_lds_b128 %0, %1, off"
                 :: "v"(lds_off), "v"((unsigned long long)(size_t)gptr)
                 : "memory");
}
__device__ __forceinline__ void wait_async_le4() {
    asm volatile("s_wait_asynccnt 0x4" ::: "memory");
}
__device__ __forceinline__ void wait_async_0() {
    asm volatile("s_wait_asynccnt 0x0" ::: "memory");
}

// ---------------- packing kernels (masks computed analytically) ----------------
__global__ __launch_bounds__(256) void pack_x_ctx(const float* __restrict__ x,
                                                  const float* __restrict__ ctx,
                                                  unsigned short* __restrict__ xb) {
    int b = blockIdx.x, t = threadIdx.x;
    float v = (t < N_IN) ? x[(size_t)b * N_IN + t] : ctx[(size_t)b * CTX + (t - N_IN)];
    xb[(size_t)b * KCAT + t] = f2bf(v);
}

__global__ __launch_bounds__(256) void pack_wcat(const float* __restrict__ w0,
                                                 const float* __restrict__ ctxw,
                                                 unsigned short* __restrict__ wcat) {
    int h = blockIdx.x, t = threadIdx.x;
    float v;
    if (t < N_IN) {
        float m = ((h % 127) >= t) ? 1.0f : 0.0f;   // deg_h >= deg_in
        v = w0[(size_t)h * N_IN + t] * m;
    } else {
        v = ctxw[(size_t)h * CTX + (t - N_IN)];
    }
    wcat[(size_t)h * KCAT + t] = f2bf(v);
}

__global__ __launch_bounds__(256) void pack_w1(const float* __restrict__ w1,
                                               unsigned short* __restrict__ dst) {
    size_t idx = (size_t)blockIdx.x * 256 + threadIdx.x;   // over H*H
    int h = (int)(idx >> 10), j = (int)(idx & 1023);
    float m = ((h % 127) >= (j % 127)) ? 1.0f : 0.0f;      // deg_h >= deg_h
    dst[idx] = f2bf(w1[idx] * m);
}

__global__ __launch_bounds__(256) void pack_w2(const float* __restrict__ w2,
                                               unsigned short* __restrict__ dst) {
    size_t idx = (size_t)blockIdx.x * 256 + threadIdx.x;   // over NOUT*H
    int o = (int)(idx >> 10), h = (int)(idx & 1023);
    float m = ((o / OUT_MULT) > (h % 127)) ? 1.0f : 0.0f;  // deg_out > deg_h
    dst[idx] = f2bf(w2[idx] * m);
}

// ---------------- WMMA GEMM:  C[M,N] = act(A[M,K] @ W[N,K]^T + bias) ----------------
// Block tile 128(M) x 128(N), KT=32. 8 wave32 waves in a 4(M) x 2(N) grid;
// each wave owns 32x64 = 2x4 v_wmma_f32_16x16x32_bf16 tiles (64 acc VGPRs).
// Global->LDS staging uses CDNA5 async copies with double buffering.
#define MT 128
#define NT 128
#define KT 32
#define LDS_STRIDE 40   // 32 + 8 pad: 16B-aligned sublines, fewer bank conflicts

__global__ __launch_bounds__(256) void gemm_wmma_bf16(
    const unsigned short* __restrict__ A,   // M x K  bf16 row-major
    const unsigned short* __restrict__ W,   // N x K  bf16 row-major (weight)
    const float*          __restrict__ bias,
    unsigned short*       __restrict__ Cb,  // bf16 out (mode 0, relu)
    float*                __restrict__ Cf,  // f32 out (mode 1, no relu)
    int M, int N, int K, int mode)
{
    __shared__ unsigned short As[2][MT][LDS_STRIDE];
    __shared__ unsigned short Ws[2][NT][LDS_STRIDE];

    const int tid   = threadIdx.x;
    const int lane  = tid & 31;
    const int wave  = tid >> 5;
    const int waveM = wave & 3;        // 4 waves over M: 32 rows each
    const int waveN = wave >> 2;       // 2 waves over N: 64 cols each
    const int mTile = blockIdx.y * MT;
    const int nTile = blockIdx.x * NT;

    const int lm = lane & 15;          // m (A) / n (B,C) within 16-tile
    const int kh = lane >> 4;          // lane-half selector per ISA layouts

    v8f acc[2][4];
    #pragma unroll
    for (int a = 0; a < 2; ++a)
        #pragma unroll
        for (int b = 0; b < 4; ++b)
            acc[a][b] = (v8f){0.f,0.f,0.f,0.f,0.f,0.f,0.f,0.f};

    const int row_ld = tid >> 1;       // 0..127 (both tiles are 128 x 32)
    const int half   = tid & 1;        // 16-bf16 half of the row

    // stage tile at K-offset kb into LDS buffer `buf` (4 async b128 per thread)
    auto stage = [&](int buf, int kb) {
        const unsigned short* ga = A + (size_t)(mTile + row_ld) * K + kb + half * 16;
        unsigned la = (unsigned)(size_t)&As[buf][row_ld][half * 16];
        async_b128(la,      ga);
        async_b128(la + 16, ga + 8);
        const unsigned short* gw = W + (size_t)(nTile + row_ld) * K + kb + half * 16;
        unsigned lw = (unsigned)(size_t)&Ws[buf][row_ld][half * 16];
        async_b128(lw,      gw);
        async_b128(lw + 16, gw + 8);
    };

    const int nK = K / KT;
    stage(0, 0);                                   // prologue

    for (int k = 0; k < nK; ++k) {
        const int cur = k & 1;
        const bool more = (k + 1 < nK);
        if (more) stage(1 - cur, (k + 1) * KT);    // overlap next tile's DMA
        if (more) wait_async_le4(); else wait_async_0();   // tile k landed (in-order)
        __syncthreads();

        // A 16x32 bf16 frag: elems 0..7 -> K = kh*8+e, elems 8..15 -> K = 16+kh*8+e
        Frag16 afr[2];
        #pragma unroll
        for (int tm = 0; tm < 2; ++tm) {
            int r = waveM * 32 + tm * 16 + lm;
            afr[tm].q[0] = *(const uint4*)&As[cur][r][kh * 8];
            afr[tm].q[1] = *(const uint4*)&As[cur][r][16 + kh * 8];
        }
        // B 32x16 bf16 frag: lane n = lm, elems 0..7 -> K = kh*8+e, elems 8..15 -> K = 16+kh*8+e
        Frag16 bfr[4];
        #pragma unroll
        for (int tn = 0; tn < 4; ++tn) {
            int r = waveN * 64 + tn * 16 + lm;
            bfr[tn].q[0] = *(const uint4*)&Ws[cur][r][kh * 8];
            bfr[tn].q[1] = *(const uint4*)&Ws[cur][r][16 + kh * 8];
        }

        #pragma unroll
        for (int tm = 0; tm < 2; ++tm)
            #pragma unroll
            for (int tn = 0; tn < 4; ++tn)
                acc[tm][tn] = __builtin_amdgcn_wmma_f32_16x16x32_bf16(
                    false, afr[tm].v, false, bfr[tn].v,
                    (short)0, acc[tm][tn], false, false);

        __syncthreads();   // all reads of buf[cur] done before k+2 overwrites it
    }

    // epilogue: C/D layout lane L -> n = L%16, VGPR r -> m = (L/16)*8 + r
    #pragma unroll
    for (int tm = 0; tm < 2; ++tm) {
        #pragma unroll
        for (int tn = 0; tn < 4; ++tn) {
            int n  = nTile + waveN * 64 + tn * 16 + lm;
            float bv = bias[n];
            int mb = mTile + waveM * 32 + tm * 16 + kh * 8;
            #pragma unroll
            for (int r = 0; r < 8; ++r) {
                float v = acc[tm][tn][r] + bv;
                if (mode == 0) {
                    v = v > 0.f ? v : 0.f;
                    Cb[(size_t)(mb + r) * N + n] = f2bf(v);
                } else {
                    Cf[(size_t)(mb + r) * N + n] = v;
                }
            }
        }
    }
}

// ---------------- rational-quadratic spline + logdet reduction ----------------
__global__ __launch_bounds__(128) void spline_kernel(const float* __restrict__ x,
                                                     const float* __restrict__ out3,
                                                     float* __restrict__ y_out,
                                                     float* __restrict__ ld_out)
{
    const float INV_SCALE = 1.0f / 724.07734393502464f;  // 1/sqrt(H*H/2)
    const int b = blockIdx.x, i = threadIdx.x;
    const float* o = out3 + (size_t)b * NOUT + i * OUT_MULT;
    const float xv = x[(size_t)b * N_IN + i];

    float uw[NB], uh[NB];
    float mw = -1e30f, mh = -1e30f;
    #pragma unroll
    for (int j = 0; j < NB; ++j) {
        uw[j] = o[j] * INV_SCALE;      mw = fmaxf(mw, uw[j]);
        uh[j] = o[NB + j] * INV_SCALE; mh = fmaxf(mh, uh[j]);
    }
    float sw = 0.f, sh = 0.f;
    #pragma unroll
    for (int j = 0; j < NB; ++j) {
        uw[j] = __expf(uw[j] - mw); sw += uw[j];
        uh[j] = __expf(uh[j] - mh); sh += uh[j];
    }
    const float rw = (1.0f - 0.001f * NB) / sw;
    const float rh = (1.0f - 0.001f * NB) / sh;

    float cw[NB + 1], ch[NB + 1], wid[NB], hei[NB];
    cw[0] = -3.f; ch[0] = -3.f;
    float aw = 0.f, ah = 0.f;
    #pragma unroll
    for (int j = 0; j < NB; ++j) {
        aw += 0.001f + rw * uw[j]; cw[j + 1] = 6.f * aw - 3.f;
        ah += 0.001f + rh * uh[j]; ch[j + 1] = 6.f * ah - 3.f;
    }
    cw[NB] = 3.f; ch[NB] = 3.f;
    #pragma unroll
    for (int j = 0; j < NB; ++j) { wid[j] = cw[j + 1] - cw[j]; hei[j] = ch[j + 1] - ch[j]; }

    float d[NB + 1];
    d[0] = 1.f; d[NB] = 1.f;   // pads: softplus(DPAD)+MIN_D == 1 exactly
    #pragma unroll
    for (int j = 0; j < NB - 1; ++j) {
        float v = o[2 * NB + j];
        float sp = (v > 20.f) ? v : log1pf(__expf(v));
        d[j + 1] = 0.001f + sp;
    }

    const bool inside = fabsf(xv) <= 3.f;
    const float xc = fminf(3.f, fmaxf(-3.f, xv));
    int idx = 0;
    #pragma unroll
    for (int j = 1; j < NB; ++j) if (xc >= cw[j]) idx = j;

    const float th    = (xc - cw[idx]) / wid[idx];
    const float delta = hei[idx] / wid[idx];
    const float d0 = d[idx], d1 = d[idx + 1];
    const float t1m   = th * (1.f - th);
    const float denom = delta + (d0 + d1 - 2.f * delta) * t1m;
    const float num   = hei[idx] * (delta * th * th + d0 * t1m);
    float yv = ch[idx] + num / denom;
    const float omt = 1.f - th;
    float ld = 2.f * __logf(delta)
             + __logf(d1 * th * th + 2.f * delta * t1m + d0 * omt * omt)
             - 2.f * __logf(denom);
    if (!inside) { yv = xv; ld = 0.f; }

    y_out[(size_t)b * N_IN + i] = yv;

    __shared__ float red[128];
    red[i] = ld; __syncthreads();
    #pragma unroll
    for (int s = 64; s > 0; s >>= 1) {
        if (i < s) red[i] += red[i + s];
        __syncthreads();
    }
    if (i == 0) ld_out[b] = red[0];
}

// ---------------- host launcher ----------------
extern "C" void kernel_launch(void* const* d_in, const int* in_sizes, int n_in,
                              void* d_out, int out_size, void* d_ws, size_t ws_size,
                              hipStream_t stream) {
    const float* x     = (const float*)d_in[0];
    const float* ctx   = (const float*)d_in[1];
    const float* ctx_w = (const float*)d_in[2];
    const float* w0    = (const float*)d_in[3];
    const float* b0    = (const float*)d_in[4];
    const float* w1    = (const float*)d_in[5];
    const float* b1    = (const float*)d_in[6];
    const float* w2    = (const float*)d_in[7];
    const float* b2    = (const float*)d_in[8];
    // d_in[9..11] = masks: recomputed analytically in the pack kernels.

    char* ws = (char*)d_ws;
    size_t off = 0;
    auto alloc = [&](size_t bytes) {
        char* p = ws + off;
        off += (bytes + 255) & ~(size_t)255;
        return p;
    };
    unsigned short* xb   = (unsigned short*)alloc((size_t)BATCH * KCAT * 2);
    unsigned short* wcat = (unsigned short*)alloc((size_t)H * KCAT * 2);
    unsigned short* w1m  = (unsigned short*)alloc((size_t)H * H * 2);
    unsigned short* w2m  = (unsigned short*)alloc((size_t)NOUT * H * 2);
    unsigned short* h1   = (unsigned short*)alloc((size_t)BATCH * H * 2);
    unsigned short* h2   = (unsigned short*)alloc((size_t)BATCH * H * 2);
    float*          out3 = (float*)alloc((size_t)BATCH * NOUT * 4);

    float* y_out  = (float*)d_out;
    float* ld_out = y_out + (size_t)BATCH * N_IN;

    pack_x_ctx<<<BATCH, 256, 0, stream>>>(x, ctx, xb);
    pack_wcat<<<H, 256, 0, stream>>>(w0, ctx_w, wcat);
    pack_w1<<<(H * H) / 256, 256, 0, stream>>>(w1, w1m);
    pack_w2<<<(NOUT * H) / 256, 256, 0, stream>>>(w2, w2m);

    // GEMM1: [16384 x 256] @ [1024 x 256]^T -> relu -> h1 (bf16)
    gemm_wmma_bf16<<<dim3(H / NT, BATCH / MT), 256, 0, stream>>>(
        xb, wcat, b0, h1, nullptr, BATCH, H, KCAT, 0);
    // GEMM2: [16384 x 1024] @ [1024 x 1024]^T -> relu -> h2 (bf16)
    gemm_wmma_bf16<<<dim3(H / NT, BATCH / MT), 256, 0, stream>>>(
        h1, w1m, b1, h2, nullptr, BATCH, H, H, 0);
    // GEMM3: [16384 x 1024] @ [2944 x 1024]^T -> out3 (f32)
    gemm_wmma_bf16<<<dim3(NOUT / NT, BATCH / MT), 256, 0, stream>>>(
        h2, w2m, b2, nullptr, out3, BATCH, NOUT, H, 1);

    spline_kernel<<<BATCH, 128, 0, stream>>>(x, out3, y_out, ld_out);
}
